// AttentionLayer_82927228552027
// MI455X (gfx1250) — matrix operand order repaired and checked by reading, not compiled, not run on hardware
//
#include <hip/hip_runtime.h>
#include <hip/hip_bf16.h>
#include <math.h>

// ---------------------------------------------------------------------------
// GAT layer for MI455X (gfx1250, wave32, WMMA).
//   N = 12288, IN = 512, OUT = 256
// Pipeline:
//   k1 gemm_h   : h = inp @ W           (split-bf16 x3 WMMA, ~fp32 accuracy)
//   k2 sv       : s1 = h@a1, s2 = h@a2  (wave reductions)
//   k3 repack   : h -> WMMA-B-swizzled bf16 hi/lo tiles (coalesced B loads)
//   k4 rowstats : per-row softmax max & 1/sum over masked leaky-relu scores
//   k5 flash    : fused  elu( softmax(mask(lrelu(s1+s2^T))) @ h )
//
// Roofline: adj scanned twice (2 x 604 MB ~ 52 us @ 23.3 TB/s HBM). h/s2/W are
// L2-resident (192 MB L2). flash uses M=32 row blocks so each block streams
// the 12.6 MB bp arrays once -> ~4.8 GB total L2 traffic, balanced against
// ~232 GFLOP of bf16 WMMA (77 GFLOP x3 split terms).
// ---------------------------------------------------------------------------

typedef __attribute__((ext_vector_type(16))) __bf16 v16bf;
typedef __attribute__((ext_vector_type(8)))  float  v8f;

#define NN   12288
#define KIN  512
#define DOUT 256
#define ALPHA 0.2f
#define NEG_BIG -9.0e15f

#define WMMA_BF16(a, b, c) \
  __builtin_amdgcn_wmma_f32_16x16x32_bf16(false, (a), false, (b), (short)0, (c), false, false)

// ------------------------------ k1: h = inp @ W ----------------------------
// One wave per 16x16 output tile. K loop in steps of 32.
__global__ __launch_bounds__(32)
void gemm_h_kernel(const float* __restrict__ inp, const float* __restrict__ W,
                   float* __restrict__ h)
{
  const int lane    = threadIdx.x & 31;
  const int n0      = blockIdx.x * 16;
  const int m0      = blockIdx.y * 16;
  const int r16     = lane & 15;
  const int hi_half = lane >> 4;       // 0 or 1
  const int kbA     = hi_half * 16;    // A-layout K base
  const int kbB     = hi_half * 16;    // B-layout K base

  v8f c = {};
  for (int kk = 0; kk < KIN; kk += 32) {
    // A tile 16x32: lane holds row r16, K = kbA+{0..15}
    const float* arow = inp + (size_t)(m0 + r16) * KIN + kk;
    float af[16];
#pragma unroll
    for (int e = 0; e < 8; ++e) { af[e] = arow[kbA + e]; af[8 + e] = arow[kbA + 8 + e]; }
    v16bf a_hi, a_lo;
#pragma unroll
    for (int e = 0; e < 16; ++e) {
      __bf16 hh = (__bf16)af[e];
      a_hi[e] = hh;
      a_lo[e] = (__bf16)(af[e] - (float)hh);
    }
    // B tile 32x16: lane holds col r16, K = kbB + e  (strided; W is L2-hot)
    const float* bcol = W + (size_t)(kk + kbB) * DOUT + n0 + r16;
    v16bf b_hi, b_lo;
#pragma unroll
    for (int e = 0; e < 16; ++e) {
      float x  = bcol[(size_t)e * DOUT];
      __bf16 hh = (__bf16)x;
      b_hi[e] = hh;
      b_lo[e] = (__bf16)(x - (float)hh);
    }
    c = WMMA_BF16(a_hi, b_hi, c);
    c = WMMA_BF16(a_lo, b_hi, c);
    c = WMMA_BF16(a_hi, b_lo, c);
  }
  // C/D layout: VGPR r -> row r (lanes 0-15) / r+8 (lanes 16-31), col = lane&15
#pragma unroll
  for (int r = 0; r < 8; ++r)
    h[(size_t)(m0 + r + hi_half * 8) * DOUT + n0 + r16] = c[r];
}

// ------------------------------ k2: s1, s2 ---------------------------------
__global__ __launch_bounds__(256)
void sv_kernel(const float* __restrict__ h, const float* __restrict__ a1,
               const float* __restrict__ a2, float* __restrict__ s1,
               float* __restrict__ s2)
{
  const int wave = threadIdx.x >> 5;
  const int lane = threadIdx.x & 31;
  const int i    = blockIdx.x * 8 + wave;
  const float* hr = h + (size_t)i * DOUT;
  float d1 = 0.f, d2 = 0.f;
#pragma unroll
  for (int e = 0; e < 8; ++e) {
    float x = hr[lane + e * 32];
    d1 += x * a1[lane + e * 32];
    d2 += x * a2[lane + e * 32];
  }
#pragma unroll
  for (int off = 16; off; off >>= 1) {
    d1 += __shfl_xor(d1, off, 32);
    d2 += __shfl_xor(d2, off, 32);
  }
  if (lane == 0) { s1[i] = d1; s2[i] = d2; }
}

// --------------------- k3: repack h into WMMA-B layout ---------------------
// Tile t = jchunk*16 + ntile is a 32x16 bf16 tile stored as [lane][16 halves]
// so the flash kernel's B load is one contiguous 32B read per lane.
__global__ __launch_bounds__(256)
void repack_kernel(const float* __restrict__ h, __bf16* __restrict__ bp_hi,
                   __bf16* __restrict__ bp_lo)
{
  const int gid  = blockIdx.x * 256 + threadIdx.x;  // tile*32 + lane
  const int lane = gid & 31;
  const int tile = gid >> 5;
  const int nt   = tile & 15;
  const int jc   = tile >> 4;
  const int col  = nt * 16 + (lane & 15);
  const int jb   = jc * 32 + (lane >> 4) * 16;      // B layout: K base per half-wave
  v16bf vhi, vlo;
#pragma unroll
  for (int e = 0; e < 16; ++e) {
    float x  = h[(size_t)(jb + e) * DOUT + col];
    __bf16 hh = (__bf16)x;
    vhi[e] = hh;
    vlo[e] = (__bf16)(x - (float)hh);
  }
  *(v16bf*)(bp_hi + (size_t)gid * 16) = vhi;
  *(v16bf*)(bp_lo + (size_t)gid * 16) = vlo;
}

// -------------------- k4: per-row softmax max & 1/sum ----------------------
__global__ __launch_bounds__(256)
void rowstats_kernel(const int* __restrict__ adj, const float* __restrict__ s1v,
                     const float* __restrict__ s2v, float* __restrict__ mrow,
                     float* __restrict__ invl)
{
  const int i    = blockIdx.x;
  const float s1 = s1v[i];
  const int* arow = adj + (size_t)i * NN;
  float m = -3.0e38f, l = 0.f;
  for (int j = threadIdx.x * 4; j < NN; j += 1024) {
    const int4   a4 = *(const int4*)(arow + j);
    const float4 t4 = *(const float4*)(s2v + j);
    const int   av[4] = { a4.x, a4.y, a4.z, a4.w };
    const float tv[4] = { t4.x, t4.y, t4.z, t4.w };
#pragma unroll
    for (int u = 0; u < 4; ++u) {
      float x  = s1 + tv[u];
      float sc = (av[u] > 0) ? (x > 0.f ? x : ALPHA * x) : NEG_BIG;
      if (sc > m) { l = l * __expf(m - sc) + 1.f; m = sc; }
      else        { l += __expf(sc - m); }
    }
  }
  __shared__ float sm[256], sl[256];
  sm[threadIdx.x] = m; sl[threadIdx.x] = l;
  __syncthreads();
  for (int off = 128; off; off >>= 1) {
    if (threadIdx.x < off) {
      float m2 = sm[threadIdx.x + off], l2 = sl[threadIdx.x + off];
      float mm = fmaxf(m, m2);
      l = l * __expf(m - mm) + l2 * __expf(m2 - mm);
      m = mm;
      sm[threadIdx.x] = m; sl[threadIdx.x] = l;
    }
    __syncthreads();
  }
  if (threadIdx.x == 0) { mrow[i] = m; invl[i] = 1.f / l; }
}

// ------------- k5: fused elu( softmax(masked scores) @ h ) -----------------
// Block = 32 output rows x 256 cols (M=32 halves B-operand L2 traffic).
// 8 waves; wave w owns cols [32w, 32w+32) for both 16-row tiles -> 4 C tiles,
// 12 WMMA per 32-wide K chunk. P tiles double-buffered: 1 barrier / chunk.
__global__ __launch_bounds__(256)
void flash_kernel(const int* __restrict__ adj, const __bf16* __restrict__ bp_hi,
                  const __bf16* __restrict__ bp_lo, const float* __restrict__ s1v,
                  const float* __restrict__ s2v, const float* __restrict__ mrow,
                  const float* __restrict__ invl, float* __restrict__ out)
{
  __shared__ __bf16 Ph[2][32][32];   // [buf][row][k]
  __shared__ __bf16 Pl[2][32][32];
  __shared__ float sS1[32], sM[32], sL[32];

  const int tid = threadIdx.x;
  const int m0  = blockIdx.x * 32;
  if (tid < 32) { sS1[tid] = s1v[m0 + tid]; sM[tid] = mrow[m0 + tid]; sL[tid] = invl[m0 + tid]; }
  __syncthreads();

  const int lane    = tid & 31;
  const int wave    = tid >> 5;      // 0..7
  const int nrow    = lane & 15;
  const int hi_half = lane >> 4;
  const int kbA     = hi_half * 16;
  const int pr      = tid >> 3;          // P row this thread fills (0..31)
  const int pc      = (tid & 7) * 4;     // 4 consecutive cols

  const float pm  = sM[pr], pil = sL[pr], ps1 = sS1[pr];
  const int* arow = adj + (size_t)(m0 + pr) * NN + pc;

  v8f c00 = {}, c01 = {}, c10 = {}, c11 = {};

  for (int jc = 0; jc < NN / 32; ++jc) {
    const int j0  = jc * 32;
    const int cur = jc & 1;

    // ---- build normalized-P tile (32x32) cooperatively in LDS[cur] ----
    {
      const int4   a4 = *(const int4*)(arow + j0);
      const float4 t4 = *(const float4*)(s2v + j0 + pc);
      if (j0 + 32 < NN) __builtin_prefetch(arow + j0 + 32, 0, 1);  // next adj chunk
      const int   av[4] = { a4.x, a4.y, a4.z, a4.w };
      const float tv[4] = { t4.x, t4.y, t4.z, t4.w };
#pragma unroll
      for (int u = 0; u < 4; ++u) {
        float x  = ps1 + tv[u];
        float sc = (av[u] > 0) ? (x > 0.f ? x : ALPHA * x) : NEG_BIG;
        float p  = __expf(sc - pm) * pil;
        __bf16 ph = (__bf16)p;
        Ph[cur][pr][pc + u] = ph;
        Pl[cur][pr][pc + u] = (__bf16)(p - (float)ph);
      }
    }
    __syncthreads();   // P[cur] visible; P[1-cur] free for next iteration

    // ---- A operands from LDS per 16-bit 16x32 layout (2 row tiles) ----
    v16bf a_hi[2], a_lo[2];
#pragma unroll
    for (int rt = 0; rt < 2; ++rt) {
      const int ar = rt * 16 + nrow;
#pragma unroll
      for (int e = 0; e < 16; ++e) {
        const int k = kbA + e;
        a_hi[rt][e] = Ph[cur][ar][k];
        a_lo[rt][e] = Pl[cur][ar][k];
      }
    }

    // ---- B operands: pre-swizzled contiguous 32B per lane per tile ----
    const size_t tbase = ((size_t)jc * 16 + wave * 2) * 32 + lane;
    const v16bf b0h = *(const v16bf*)(bp_hi + tbase * 16);
    const v16bf b0l = *(const v16bf*)(bp_lo + tbase * 16);
    const v16bf b1h = *(const v16bf*)(bp_hi + (tbase + 32) * 16);
    const v16bf b1l = *(const v16bf*)(bp_lo + (tbase + 32) * 16);

    c00 = WMMA_BF16(a_hi[0], b0h, c00);
    c00 = WMMA_BF16(a_lo[0], b0h, c00);
    c00 = WMMA_BF16(a_hi[0], b0l, c00);
    c01 = WMMA_BF16(a_hi[0], b1h, c01);
    c01 = WMMA_BF16(a_lo[0], b1h, c01);
    c01 = WMMA_BF16(a_hi[0], b1l, c01);
    c10 = WMMA_BF16(a_hi[1], b0h, c10);
    c10 = WMMA_BF16(a_lo[1], b0h, c10);
    c10 = WMMA_BF16(a_hi[1], b0l, c10);
    c11 = WMMA_BF16(a_hi[1], b1h, c11);
    c11 = WMMA_BF16(a_lo[1], b1h, c11);
    c11 = WMMA_BF16(a_hi[1], b1l, c11);
  }

  // ---- epilogue: ELU + store (4 tiles) ----
  const int n0 = wave * 32;
#pragma unroll
  for (int r = 0; r < 8; ++r) {
    const int mA = m0 + r + hi_half * 8;        // row tile 0
    const int mB = mA + 16;                     // row tile 1
    float x00 = c00[r], x01 = c01[r], x10 = c10[r], x11 = c11[r];
    x00 = x00 > 0.f ? x00 : expm1f(x00);
    x01 = x01 > 0.f ? x01 : expm1f(x01);
    x10 = x10 > 0.f ? x10 : expm1f(x10);
    x11 = x11 > 0.f ? x11 : expm1f(x11);
    out[(size_t)mA * DOUT + n0 + nrow]      = x00;
    out[(size_t)mA * DOUT + n0 + 16 + nrow] = x01;
    out[(size_t)mB * DOUT + n0 + nrow]      = x10;
    out[(size_t)mB * DOUT + n0 + 16 + nrow] = x11;
  }
}

// ------------------------------ launcher -----------------------------------
extern "C" void kernel_launch(void* const* d_in, const int* in_sizes, int n_in,
                              void* d_out, int out_size, void* d_ws, size_t ws_size,
                              hipStream_t stream)
{
  (void)in_sizes; (void)n_in; (void)out_size; (void)ws_size;
  const float* inp = (const float*)d_in[0];
  const int*   adj = (const int*)d_in[1];
  const float* W   = (const float*)d_in[2];
  const float* a1  = (const float*)d_in[3];
  const float* a2  = (const float*)d_in[4];
  float* out = (float*)d_out;

  char* ws = (char*)d_ws;
  size_t off = 0;
  float*  h    = (float*)(ws + off);  off += (size_t)NN * DOUT * sizeof(float);   // 12.6 MB
  __bf16* bph  = (__bf16*)(ws + off); off += (size_t)NN * DOUT * sizeof(__bf16);  // 6.3 MB
  __bf16* bpl  = (__bf16*)(ws + off); off += (size_t)NN * DOUT * sizeof(__bf16);  // 6.3 MB
  float*  s1   = (float*)(ws + off);  off += (size_t)NN * sizeof(float);
  float*  s2   = (float*)(ws + off);  off += (size_t)NN * sizeof(float);
  float*  mrow = (float*)(ws + off);  off += (size_t)NN * sizeof(float);
  float*  invl = (float*)(ws + off);  off += (size_t)NN * sizeof(float);

  gemm_h_kernel  <<<dim3(DOUT / 16, NN / 16), 32,  0, stream>>>(inp, W, h);
  sv_kernel      <<<NN / 8,                    256, 0, stream>>>(h, a1, a2, s1, s2);
  repack_kernel  <<<(NN / 32) * 16 * 32 / 256, 256, 0, stream>>>(h, bph, bpl);
  rowstats_kernel<<<NN,                        256, 0, stream>>>(adj, s1, s2, mrow, invl);
  flash_kernel   <<<NN / 32,                   256, 0, stream>>>(adj, bph, bpl, s1, s2,
                                                                 mrow, invl, out);
}
